// MML_75316546503120
// MI455X (gfx1250) — matrix operand order repaired and checked by reading, not compiled, not run on hardware
//
#include <hip/hip_runtime.h>

typedef _Float16 h16;
typedef __attribute__((ext_vector_type(16))) _Float16 v16h;
typedef __attribute__((ext_vector_type(8)))  _Float16 v8h;
typedef __attribute__((ext_vector_type(8)))  float    v8f;

union V16U { v16h v; v8h h[2]; };

// ---- WMMA fragment loaders (ISA 7.12.2 layouts, wave32) --------------------
// A (16x32 f16): row = lane&15; lanes<16 hold K {kb..kb+7, kb+16..kb+23},
// lanes>=16 hold K {kb+8..kb+15, kb+24..kb+31}.
static __device__ inline v16h frag_a(const h16* arow, int kb, int lane) {
    int off = (lane >= 16) ? 8 : 0;
    V16U u;
    u.h[0] = *(const v8h*)(arow + kb + off);
    u.h[1] = *(const v8h*)(arow + kb + 16 + off);
    return u.v;
}
// B (32x16 f16): col = lane&15; lanes<16 hold K {kb..kb+7, kb+16..kb+23},
// lanes>=16 hold K {kb+8..kb+15, kb+24..kb+31}.
static __device__ inline v16h frag_b(const h16* bcol, int kb, int lane) {
    int off = (lane >= 16) ? 8 : 0;
    V16U u;
    u.h[0] = *(const v8h*)(bcol + kb + off);
    u.h[1] = *(const v8h*)(bcol + kb + off + 16);
    return u.v;
}

// ---- fused GEMM + (max over cols, sum over row groups) ---------------------
// A: [batch][rowsPad][K] f16 row-major.  B: [(b*5+n)][colsPad][K] f16 (col-major
// of the logical [K,cols] matrix).  For each row: max over valid cols, then
// atomicAdd into sim[(b*Q + row/row_div)*5 + n].
// col_tiles must be even; each wave processes 2 adjacent tiles per pass
// (shared A fragment, 2 independent accumulators for WMMA ILP).
__global__ void __launch_bounds__(256)
k_gemm_reduce(const h16* __restrict__ A, const h16* __restrict__ B,
              float* __restrict__ sim, int rowsPad, int rows_real, int row_div,
              int cols_pad, int cols_real, int col_tiles, int K)
{
    __shared__ __align__(16) h16 ldsA[16 * 640];
    __shared__ float wmax[8][16];
    const int rt = blockIdx.x, n = blockIdx.y, b = blockIdx.z;
    const int tid = threadIdx.x, wave = tid >> 5, lane = tid & 31;
    const int row0 = rt * 16;

    const h16* Ab = A + (size_t)b * rowsPad * K;
    const int kchunks = K >> 3;           // v8h chunks per row
    for (int ch = tid; ch < 16 * kchunks; ch += 256) {
        int r = ch / kchunks, kk = (ch % kchunks) << 3;
        *(v8h*)(ldsA + r * K + kk) = *(const v8h*)(Ab + (size_t)(row0 + r) * K + kk);
    }
    __syncthreads();

    const h16* Bb = B + (size_t)(b * 5 + n) * cols_pad * K;
    const h16* arow = ldsA + (size_t)(lane & 15) * K;

    float rmax[8];
#pragma unroll
    for (int r = 0; r < 8; ++r) rmax[r] = -3.0e38f;

    for (int t = 2 * wave; t < col_tiles; t += 16) {
        const int col0 = t * 16 + (lane & 15);
        const h16* bcol0 = Bb + (size_t)col0 * K;
        const h16* bcol1 = bcol0 + (size_t)16 * K;
        v8f acc0 = {}, acc1 = {};
        for (int kb = 0; kb < K; kb += 32) {
            v16h a  = frag_a(arow, kb, lane);
            v16h b0 = frag_b(bcol0, kb, lane);
            v16h b1 = frag_b(bcol1, kb, lane);
            acc0 = __builtin_amdgcn_wmma_f32_16x16x32_f16(
                false, a, false, b0, (short)0, acc0, false, false);
            acc1 = __builtin_amdgcn_wmma_f32_16x16x32_f16(
                false, a, false, b1, (short)0, acc1, false, false);
        }
        if (col0 < cols_real) {
#pragma unroll
            for (int r = 0; r < 8; ++r) rmax[r] = fmaxf(rmax[r], acc0[r]);
        }
        if (col0 + 16 < cols_real) {
#pragma unroll
            for (int r = 0; r < 8; ++r) rmax[r] = fmaxf(rmax[r], acc1[r]);
        }
    }
    // max across the 16 lanes of each half-wave (same row, different cols)
#pragma unroll
    for (int d = 1; d < 16; d <<= 1) {
#pragma unroll
        for (int r = 0; r < 8; ++r) rmax[r] = fmaxf(rmax[r], __shfl_xor(rmax[r], d, 32));
    }
    if ((lane & 15) == 0) {
        const int half = lane >> 4;
#pragma unroll
        for (int r = 0; r < 8; ++r) wmax[wave][r + 8 * half] = rmax[r];
    }
    __syncthreads();
    if (tid < 16) {
        float m = -3.0e38f;
#pragma unroll
        for (int w = 0; w < 8; ++w) m = fmaxf(m, wmax[w][tid]);
        const int row = row0 + tid;
        if (row < rows_real) {
            const int Q = rows_real / row_div;
            atomicAdd(&sim[((size_t)b * Q + row / row_div) * 5 + n], m);
        }
    }
}

// ---- generic f16 WMMA GEMM: C = A * B^T (B stored [col][k]) ----------------
// Each wave computes 2 adjacent 16x16 tiles (shared A fragment, 2 accumulators).
__global__ void __launch_bounds__(128)
k_gemm16(const h16* __restrict__ A, const h16* __restrict__ B,
         float* __restrict__ Cf, h16* __restrict__ Ch,
         int n_tiles, int K, long strideA, long strideB, long strideC, int ldc)
{
    __shared__ __align__(16) h16 ldsA[16 * 640];
    const int tid = threadIdx.x, wave = tid >> 5, lane = tid & 31;
    const int b = blockIdx.z;
    const int row0 = blockIdx.x * 16;

    const h16* Ab = A + (size_t)b * strideA;
    const int kchunks = K >> 3;
    for (int ch = tid; ch < 16 * kchunks; ch += 128) {
        int r = ch / kchunks, kk = (ch % kchunks) << 3;
        *(v8h*)(ldsA + r * K + kk) = *(const v8h*)(Ab + (size_t)(row0 + r) * K + kk);
    }
    __syncthreads();

    const int t0 = (blockIdx.y * 4 + wave) * 2;
    if (t0 >= n_tiles) return;
    const bool has1 = (t0 + 1) < n_tiles;
    const int col0 = t0 * 16 + (lane & 15);
    const h16* bcol0 = B + (size_t)b * strideB + (size_t)col0 * K;
    const h16* bcol1 = has1 ? bcol0 + (size_t)16 * K : bcol0;
    const h16* arow = ldsA + (size_t)(lane & 15) * K;

    v8f acc0 = {}, acc1 = {};
    for (int kb = 0; kb < K; kb += 32) {
        v16h a  = frag_a(arow, kb, lane);
        v16h b0 = frag_b(bcol0, kb, lane);
        v16h b1 = frag_b(bcol1, kb, lane);
        acc0 = __builtin_amdgcn_wmma_f32_16x16x32_f16(
            false, a, false, b0, (short)0, acc0, false, false);
        acc1 = __builtin_amdgcn_wmma_f32_16x16x32_f16(
            false, a, false, b1, (short)0, acc1, false, false);
    }
    const int rbase = row0 + 8 * (lane >> 4);
    float* cf = Cf ? Cf + (size_t)b * strideC : nullptr;
    h16* chp = Ch ? Ch + (size_t)b * strideC : nullptr;
#pragma unroll
    for (int r = 0; r < 8; ++r) {
        const size_t idx = (size_t)(rbase + r) * ldc + col0;
        if (cf) {
            cf[idx] = acc0[r];
            if (has1) cf[idx + 16] = acc1[r];
        }
        if (chp) {
            chp[idx] = (h16)acc0[r];
            if (has1) chp[idx + 16] = (h16)acc1[r];
        }
    }
}

// ---- preprocessing / small kernels -----------------------------------------
// qf16[b][row=q*100+m][c] = l2norm_c(query_xf[b][q][c][m]), f16, rowsPad=7504
__global__ void k_norm_q_pixels(const float* __restrict__ q, h16* __restrict__ out) {
    const int gw = (blockIdx.x * blockDim.x + threadIdx.x) >> 5;
    const int lane = threadIdx.x & 31;
    if (gw >= 4 * 7500) return;
    const int b = gw / 7500, row = gw % 7500;
    const int qq = row / 100, m = row % 100;
    const float* src = q + ((size_t)(b * 75 + qq) * 640) * 100 + m;
    float ss = 0.f;
    for (int c = lane; c < 640; c += 32) { float v = src[(size_t)c * 100]; ss += v * v; }
    for (int d = 1; d < 32; d <<= 1) ss += __shfl_xor(ss, d, 32);
    const float rn = 1.0f / (sqrtf(ss) + 1e-12f);
    h16* dst = out + ((size_t)b * 7504 + row) * 640;
    for (int c = lane; c < 640; c += 32) dst[c] = (h16)(src[(size_t)c * 100] * rn);
}

// sf16[(b*5+n)][col=ks*100+m][c] = l2norm_c(support_xf[b][n*5+ks][c][m]), colsPad=512
__global__ void k_norm_s_pixels(const float* __restrict__ s, h16* __restrict__ out) {
    const int gw = (blockIdx.x * blockDim.x + threadIdx.x) >> 5;
    const int lane = threadIdx.x & 31;
    if (gw >= 4 * 5 * 500) return;
    const int b = gw / 2500, rem = gw % 2500, n = rem / 500, col = rem % 500;
    const int ks = col / 100, m = col % 100;
    const float* src = s + ((size_t)(b * 25 + n * 5 + ks) * 640) * 100 + m;
    float ss = 0.f;
    for (int c = lane; c < 640; c += 32) { float v = src[(size_t)c * 100]; ss += v * v; }
    for (int d = 1; d < 32; d <<= 1) ss += __shfl_xor(ss, d, 32);
    const float rn = 1.0f / (sqrtf(ss) + 1e-12f);
    h16* dst = out + ((size_t)(b * 5 + n) * 512 + col) * 640;
    for (int c = lane; c < 640; c += 32) dst[c] = (h16)(src[(size_t)c * 100] * rn);
}

// qp16[b][row=q*13+p][c] = l2norm_c(query_xf_part[b][q][c][p]), rowsPad=976
__global__ void k_norm_qp(const float* __restrict__ q, h16* __restrict__ out) {
    const int gw = (blockIdx.x * blockDim.x + threadIdx.x) >> 5;
    const int lane = threadIdx.x & 31;
    if (gw >= 4 * 975) return;
    const int b = gw / 975, row = gw % 975;
    const int qq = row / 13, p = row % 13;
    const float* src = q + ((size_t)(b * 75 + qq) * 640) * 13 + p;
    float ss = 0.f;
    for (int c = lane; c < 640; c += 32) { float v = src[(size_t)c * 13]; ss += v * v; }
    for (int d = 1; d < 32; d <<= 1) ss += __shfl_xor(ss, d, 32);
    const float rn = 1.0f / (sqrtf(ss) + 1e-12f);
    h16* dst = out + ((size_t)b * 976 + row) * 640;
    for (int c = lane; c < 640; c += 32) dst[c] = (h16)(src[(size_t)c * 13] * rn);
}

// sp_raw[b][node][c] (f32) and sp16 (f16) from support_xf_part; node = sl*13+p
__global__ void k_build_sp(const float* __restrict__ src, float* __restrict__ spr,
                           h16* __restrict__ sph) {
    const size_t i = (size_t)blockIdx.x * blockDim.x + threadIdx.x;
    if (i >= (size_t)4 * 325 * 640) return;
    const int b = (int)(i / (325 * 640)), rem = (int)(i % (325 * 640));
    const int node = rem / 640, c = rem % 640;
    const int sl = node / 13, p = node % 13;
    const float v = src[((size_t)(b * 25 + sl) * 640 + c) * 13 + p];
    spr[((size_t)b * 336 + node) * 640 + c] = v;
    sph[((size_t)b * 336 + node) * 640 + c] = (h16)v;
}

// w16[w][co][ci] = W_w[ci][co]  (B layout: [out col][in k])
__global__ void k_w16(const float* __restrict__ Wq, const float* __restrict__ Wk,
                      const float* __restrict__ Wv, h16* __restrict__ out) {
    const size_t i = (size_t)blockIdx.x * blockDim.x + threadIdx.x;
    if (i >= (size_t)3 * 640 * 640) return;
    const int w = (int)(i / (640 * 640)), rem = (int)(i % (640 * 640));
    const int co = rem / 640, ci = rem % 640;
    const float* W = (w == 0) ? Wq : (w == 1) ? Wk : Wv;
    out[(size_t)w * 640 * 640 + (size_t)co * 640 + ci] = (h16)W[(size_t)ci * 640 + co];
}

// row-wise softmax of S/sqrt(c) over j<325 -> P16 [336][352] f16 (pads stay 0)
__global__ void k_softmax(const float* __restrict__ S, h16* __restrict__ P) {
    const int gw = (blockIdx.x * blockDim.x + threadIdx.x) >> 5;
    const int lane = threadIdx.x & 31;
    if (gw >= 4 * 325) return;
    const int b = gw / 325, i = gw % 325;
    const float* row = S + ((size_t)b * 336 + i) * 336;
    const float sc = 0.0395284708f; // 1/sqrt(640)
    float m = -3.0e38f;
    for (int j = lane; j < 325; j += 32) m = fmaxf(m, row[j] * sc);
    for (int d = 1; d < 32; d <<= 1) m = fmaxf(m, __shfl_xor(m, d, 32));
    float s = 0.f;
    for (int j = lane; j < 325; j += 32) s += __expf(row[j] * sc - m);
    for (int d = 1; d < 32; d <<= 1) s += __shfl_xor(s, d, 32);
    const float inv = 1.0f / s;
    h16* pr = P + ((size_t)b * 336 + i) * 352;
    for (int j = lane; j < 325; j += 32) pr[j] = (h16)(__expf(row[j] * sc - m) * inv);
}

// V16t[b][c][j] = V16[b][j][c]  (B layout for P@V, K padded to 352)
__global__ void k_vt(const h16* __restrict__ V, h16* __restrict__ Vt) {
    const size_t i = (size_t)blockIdx.x * blockDim.x + threadIdx.x;
    if (i >= (size_t)4 * 640 * 325) return;
    const int b = (int)(i / (640 * 325)), rem = (int)(i % (640 * 325));
    const int c = rem / 325, j = rem % 325;
    Vt[((size_t)b * 640 + c) * 352 + j] = V[((size_t)b * 336 + j) * 640 + c];
}

// spB16[(b*5+n)][col=ks*13+p][c] = l2norm_c(sp_raw + O), colsPad=96
__global__ void k_build_spB(const float* __restrict__ spr, const float* __restrict__ O,
                            h16* __restrict__ out) {
    const int gw = (blockIdx.x * blockDim.x + threadIdx.x) >> 5;
    const int lane = threadIdx.x & 31;
    if (gw >= 4 * 5 * 65) return;
    const int b = gw / 325, rem = gw % 325, n = rem / 65, col = rem % 65;
    const int node = (n * 5 + col / 13) * 13 + (col % 13);
    const float* a = spr + ((size_t)b * 336 + node) * 640;
    const float* o = O + ((size_t)b * 336 + node) * 640;
    float ss = 0.f;
    for (int c = lane; c < 640; c += 32) { float v = a[c] + o[c]; ss += v * v; }
    for (int d = 1; d < 32; d <<= 1) ss += __shfl_xor(ss, d, 32);
    const float rn = 1.0f / (sqrtf(ss) + 1e-12f);
    h16* dst = out + ((size_t)(b * 5 + n) * 96 + col) * 640;
    for (int c = lane; c < 640; c += 32) dst[c] = (h16)((a[c] + o[c]) * rn);
}

__global__ void k_proto(const float* __restrict__ s, float* __restrict__ proto) {
    const int i = blockIdx.x * blockDim.x + threadIdx.x;
    if (i >= 4 * 5 * 640) return;
    const int b = i / 3200, rem = i % 3200, n = rem / 640, c = rem % 640;
    float acc = 0.f;
    for (int ks = 0; ks < 5; ++ks) {
        const float* p = s + ((size_t)(b * 25 + n * 5 + ks) * 640 + c) * 100;
        for (int m = 0; m < 100; ++m) acc += p[m];
    }
    proto[i] = acc * (1.0f / 500.0f);
}

__global__ void k_qg(const float* __restrict__ q, float* __restrict__ qg) {
    const int i = blockIdx.x * blockDim.x + threadIdx.x;
    if (i >= 4 * 75 * 640) return;
    const float* p = q + (size_t)i * 100;
    float acc = 0.f;
    for (int m = 0; m < 100; ++m) acc += p[m];
    qg[i] = acc * 0.01f;
}

__global__ void k_gsim(const float* __restrict__ qg, const float* __restrict__ proto,
                       float* __restrict__ gs) {
    const int i = blockIdx.x * blockDim.x + threadIdx.x;
    if (i >= 4 * 75 * 5) return;
    const int b = i / 375, rem = i % 375, qq = rem / 5, n = rem % 5;
    const float* a = qg + ((size_t)b * 75 + qq) * 640;
    const float* p = proto + ((size_t)b * 5 + n) * 640;
    float d2 = 0.f;
    for (int c = 0; c < 640; ++c) { float d = a[c] - p[c]; d2 += d * d; }
    gs[((size_t)b * 75 + qq) * 5 + n] = -d2;
}

__global__ void k_combine(const float* __restrict__ part, const float* __restrict__ pix,
                          const float* __restrict__ glob, float* __restrict__ out) {
    const int r = blockIdx.x * blockDim.x + threadIdx.x;
    if (r >= 300) return;
    float o[5] = {0.f, 0.f, 0.f, 0.f, 0.f};
    const float* srcs[3] = {part, pix, glob};
    const float wgt[3] = {1.0f, 0.5f, 0.5f};
    for (int s3 = 0; s3 < 3; ++s3) {
        float v[5], m = -3.0e38f, sum = 0.f;
        for (int n = 0; n < 5; ++n) { v[n] = srcs[s3][r * 5 + n]; m = fmaxf(m, v[n]); }
        for (int n = 0; n < 5; ++n) { v[n] = __expf(v[n] - m); sum += v[n]; }
        const float inv = wgt[s3] / sum;
        for (int n = 0; n < 5; ++n) o[n] += v[n] * inv;
    }
    for (int n = 0; n < 5; ++n) out[r * 5 + n] = o[n];
}

extern "C" void kernel_launch(void* const* d_in, const int* in_sizes, int n_in,
                              void* d_out, int out_size, void* d_ws, size_t ws_size,
                              hipStream_t stream) {
    (void)in_sizes; (void)n_in; (void)out_size; (void)ws_size;
    const float* support_xf   = (const float*)d_in[0];
    const float* query_xf     = (const float*)d_in[1];
    const float* support_part = (const float*)d_in[2];
    const float* query_part   = (const float*)d_in[3];
    const float* Wq = (const float*)d_in[4];
    const float* Wk = (const float*)d_in[5];
    const float* Wv = (const float*)d_in[6];
    float* out = (float*)d_out;

    uint8_t* base = (uint8_t*)d_ws;
    size_t off = 0;
    auto alloc = [&](size_t bytes) -> void* {
        void* p = base + off; off += (bytes + 255) & ~(size_t)255; return p;
    };
    h16*   qf16   = (h16*)  alloc((size_t)4 * 7504 * 640 * 2);
    h16*   sf16   = (h16*)  alloc((size_t)4 * 5 * 512 * 640 * 2);
    h16*   qp16   = (h16*)  alloc((size_t)4 * 976 * 640 * 2);
    h16*   spB16  = (h16*)  alloc((size_t)4 * 5 * 96 * 640 * 2);
    float* sp_raw = (float*)alloc((size_t)4 * 336 * 640 * 4);
    h16*   sp16   = (h16*)  alloc((size_t)4 * 336 * 640 * 2);
    h16*   w16    = (h16*)  alloc((size_t)3 * 640 * 640 * 2);
    h16*   Q16    = (h16*)  alloc((size_t)4 * 336 * 640 * 2);
    h16*   K16m   = (h16*)  alloc((size_t)4 * 336 * 640 * 2);
    h16*   V16m   = (h16*)  alloc((size_t)4 * 336 * 640 * 2);
    h16*   V16t   = (h16*)  alloc((size_t)4 * 640 * 352 * 2);
    float* Sf     = (float*)alloc((size_t)4 * 336 * 336 * 4);
    h16*   P16    = (h16*)  alloc((size_t)4 * 336 * 352 * 2);
    float* Of     = (float*)alloc((size_t)4 * 336 * 640 * 4);
    float* proto  = (float*)alloc((size_t)4 * 5 * 640 * 4);
    float* qg     = (float*)alloc((size_t)4 * 75 * 640 * 4);
    float* pixs   = (float*)alloc((size_t)300 * 5 * 4);
    float* parts  = (float*)alloc((size_t)300 * 5 * 4);
    float* globs  = (float*)alloc((size_t)300 * 5 * 4);

    // zero workspace: pads must be 0, sims accumulate via atomics
    hipMemsetAsync(d_ws, 0, off, stream);

    k_w16<<<(3 * 640 * 640 + 255) / 256, 256, 0, stream>>>(Wq, Wk, Wv, w16);
    k_build_sp<<<(4 * 325 * 640 + 255) / 256, 256, 0, stream>>>(support_part, sp_raw, sp16);
    k_norm_q_pixels<<<(4 * 7500 * 32 + 255) / 256, 256, 0, stream>>>(query_xf, qf16);
    k_norm_s_pixels<<<(4 * 5 * 500 * 32 + 255) / 256, 256, 0, stream>>>(support_xf, sf16);
    k_norm_qp<<<(4 * 975 * 32 + 255) / 256, 256, 0, stream>>>(query_part, qp16);
    k_proto<<<(4 * 5 * 640 + 255) / 256, 256, 0, stream>>>(support_xf, proto);
    k_qg<<<(4 * 75 * 640 + 255) / 256, 256, 0, stream>>>(query_xf, qg);
    k_gsim<<<(1500 + 255) / 256, 256, 0, stream>>>(qg, proto, globs);

    // attention: QKV projections (A = sp16 [336x640], B = weight [640 cols][640 k])
    // 4 waves x 2 tiles = 8 col-tiles per block; 40 tiles -> grid.y = 5
    dim3 gQKV(21, 5, 4);
    k_gemm16<<<gQKV, 128, 0, stream>>>(sp16, w16,                  nullptr, Q16, 40, 640,
                                       336L * 640, 0, 336L * 640, 640);
    k_gemm16<<<gQKV, 128, 0, stream>>>(sp16, w16 + 640L * 640,     nullptr, K16m, 40, 640,
                                       336L * 640, 0, 336L * 640, 640);
    k_gemm16<<<gQKV, 128, 0, stream>>>(sp16, w16 + 2L * 640 * 640, nullptr, V16m, 40, 640,
                                       336L * 640, 0, 336L * 640, 640);
    // S = Q K^T : 21 col-tiles -> grid.y = 3
    dim3 gS(21, 3, 4);
    k_gemm16<<<gS, 128, 0, stream>>>(Q16, K16m, Sf, nullptr, 21, 640,
                                     336L * 640, 336L * 640, 336L * 336, 336);
    k_softmax<<<(4 * 325 * 32 + 255) / 256, 256, 0, stream>>>(Sf, P16);
    k_vt<<<(4 * 640 * 325 + 255) / 256, 256, 0, stream>>>(V16m, V16t);
    // O = P V : 40 col-tiles -> grid.y = 5
    dim3 gO(21, 5, 4);
    k_gemm16<<<gO, 128, 0, stream>>>(P16, V16t, Of, nullptr, 40, 352,
                                     336L * 352, 640L * 352, 336L * 640, 640);
    k_build_spB<<<(4 * 5 * 65 * 32 + 255) / 256, 256, 0, stream>>>(sp_raw, Of, spB16);

    // fused pixel NBNN: [7504x640] x [512x640]^T per (b,n), max over s, sum over m
    dim3 gPix(469, 5, 4);
    k_gemm_reduce<<<gPix, 256, 0, stream>>>(qf16, sf16, pixs,
                                            7504, 7500, 100, 512, 500, 32, 640);
    // fused part similarity: [976x640] x [96x640]^T per (b,n), 6 even col-tiles
    dim3 gPart(61, 5, 4);
    k_gemm_reduce<<<gPart, 256, 0, stream>>>(qp16, spB16, parts,
                                             976, 975, 13, 96, 65, 6, 640);

    k_combine<<<2, 256, 0, stream>>>(parts, pixs, globs, out);
}
